// Attention_59665685676311
// MI455X (gfx1250) — compile-verified, matrix operand order checked by reading, not run on hardware
//
#include <hip/hip_runtime.h>
#include <hip/hip_bf16.h>

typedef __bf16 bf16_t;
typedef __attribute__((ext_vector_type(8)))  bf16_t v8bf;
typedef __attribute__((ext_vector_type(16))) bf16_t v16bf;
typedef __attribute__((ext_vector_type(8)))  float  v8f;
typedef __attribute__((ext_vector_type(4)))  int    v4i;

#define D_MODEL 1024
#define NH      16
#define HD      64
#define SEQ     2048
#define BATCH   4
#define ROWS    (BATCH*SEQ)   /* 8192 */
#define LDA     40            /* GEMM LDS stride (bf16), 80B = 16B-aligned */
#define LDK     72            /* attention LDS stride (bf16), 144B = 16B-aligned */

#if __has_builtin(__builtin_amdgcn_global_load_async_to_lds_b128)
#define HAVE_ASYNC 1
#define ASYNC_CP_B128(dst, src)                                         \
  __builtin_amdgcn_global_load_async_to_lds_b128(                       \
      (__attribute__((address_space(1))) v4i*)(src),                    \
      (__attribute__((address_space(3))) v4i*)(dst), 0, 0)
#define ASYNC_WAIT() asm volatile("s_wait_asynccnt 0" ::: "memory")
#else
#define HAVE_ASYNC 0
#endif

static __device__ __forceinline__ v16bf join16(v8bf lo, v8bf hi) {
  return __builtin_shufflevector(lo, hi, 0,1,2,3,4,5,6,7,8,9,10,11,12,13,14,15);
}
static __device__ __forceinline__ v8f wmma_bf16(v16bf a, v16bf b, v8f c) {
  return __builtin_amdgcn_wmma_f32_16x16x32_bf16(false, a, false, b, (short)0, c, false, false);
}
static __device__ __forceinline__ v16bf frag_from(const bf16_t* row, int o0, int o1) {
  return join16(*(const v8bf*)(row + o0), *(const v8bf*)(row + o1));
}

// ---------------------------------------------------------------------------
// Kernel 1: qkv = x @ w_qkv ; scatter to head-major bf16 Q (pre-scaled), K, V
// Block tile 256x128, 8 waves, wave tile 32x128 (2x8 fragments, 16 WMMA/k-step)
// ---------------------------------------------------------------------------
__global__ __launch_bounds__(256)
void qkv_kernel(const float* __restrict__ x, const float* __restrict__ w,
                bf16_t* __restrict__ Qh, bf16_t* __restrict__ Kh, bf16_t* __restrict__ Vh)
{
  __shared__ __align__(16) bf16_t As[256*LDA];
  __shared__ __align__(16) bf16_t Bs[128*LDA];
  const int tid = threadIdx.x, wave = tid >> 5, lane = tid & 31;
  const int l15 = lane & 15, g = lane >> 4;
  const int m0 = blockIdx.x * 256, n0 = blockIdx.y * 128;

  v8f acc[2][8];
  #pragma unroll
  for (int i = 0; i < 2; ++i)
    #pragma unroll
    for (int j = 0; j < 8; ++j) acc[i][j] = (v8f){0,0,0,0,0,0,0,0};

  for (int k0 = 0; k0 < D_MODEL; k0 += 32) {
    #pragma unroll
    for (int c = 0; c < 8; ++c) {                    // A: 256x32 fp32 -> bf16
      int linear = tid + 256*c;
      int row = linear >> 3, kk = (linear & 7) << 2;
      const float4 f = *(const float4*)(x + (size_t)(m0+row)*D_MODEL + k0 + kk);
      bf16_t* p = As + row*LDA + kk;
      p[0]=(bf16_t)f.x; p[1]=(bf16_t)f.y; p[2]=(bf16_t)f.z; p[3]=(bf16_t)f.w;
    }
    #pragma unroll
    for (int c = 0; c < 4; ++c) {                    // B: 32x128 -> col-major bf16
      int linear = tid + 256*c;
      int kk = linear >> 5, col = (linear & 31) << 2;
      const float4 f = *(const float4*)(w + (size_t)(k0+kk)*3072 + n0 + col);
      Bs[(col+0)*LDA + kk] = (bf16_t)f.x;
      Bs[(col+1)*LDA + kk] = (bf16_t)f.y;
      Bs[(col+2)*LDA + kk] = (bf16_t)f.z;
      Bs[(col+3)*LDA + kk] = (bf16_t)f.w;
    }
    __syncthreads();
    const bf16_t* arow0 = As + (wave*32 + l15)*LDA;
    v16bf a0 = frag_from(arow0,          g*8, 16 + g*8);
    v16bf a1 = frag_from(arow0 + 16*LDA, g*8, 16 + g*8);
    #pragma unroll
    for (int j = 0; j < 8; ++j) {
      const bf16_t* brow = Bs + (j*16 + l15)*LDA;
      v16bf bfrag = frag_from(brow, g*8, 16 + g*8);
      acc[0][j] = wmma_bf16(a0, bfrag, acc[0][j]);
      acc[1][j] = wmma_bf16(a1, bfrag, acc[1][j]);
    }
    __syncthreads();
  }

  const int region = n0 >> 10;                       // 0=Q 1=K 2=V
  bf16_t* base = (region == 0) ? Qh : (region == 1) ? Kh : Vh;
  const float scl = (region == 0) ? 0.125f : 1.0f;   // hd^-0.5 folded into Q
  const int ncol0 = n0 & (D_MODEL-1);
  #pragma unroll
  for (int i = 0; i < 2; ++i)
    #pragma unroll
    for (int j = 0; j < 8; ++j) {
      int col = ncol0 + j*16 + l15;
      int h = col >> 6, hd = col & 63;
      #pragma unroll
      for (int r = 0; r < 8; ++r) {
        int row = m0 + wave*32 + i*16 + r + 8*g;
        int b = row >> 11, n = row & (SEQ-1);
        base[(((size_t)(b*NH + h))*SEQ + n)*HD + hd] = (bf16_t)(acc[i][j][r] * scl);
      }
    }
}

// ---------------------------------------------------------------------------
// Kernel 2: flash attention per (b,h); 64 query rows per block, 64-key tiles
// ---------------------------------------------------------------------------
__global__ __launch_bounds__(128)
void attn_kernel(const bf16_t* __restrict__ Qh, const bf16_t* __restrict__ Kh,
                 const bf16_t* __restrict__ Vh, bf16_t* __restrict__ O)
{
  __shared__ __align__(16) bf16_t Ks [64*LDK];       // [key][hd]
  __shared__ __align__(16) bf16_t Vst[HD*LDK];       // [hd][key]  (transposed)
  __shared__ __align__(16) bf16_t Ps [4*16*LDK];     // per-wave P staging

  const int tid = threadIdx.x, wave = tid >> 5, lane = tid & 31;
  const int l15 = lane & 15, g = lane >> 4;
  const int bh = blockIdx.x, q0 = blockIdx.y * 64;
  const int b = bh >> 4, h = bh & 15;
  const size_t headoff = (size_t)bh * SEQ * HD;

  const bf16_t* qrow = Qh + headoff + (size_t)(q0 + wave*16 + l15)*HD;
  v16bf aq0 = frag_from(qrow,      g*8, 16 + g*8);
  v16bf aq1 = frag_from(qrow, 32 + g*8, 48 + g*8);

  v8f oacc[4];
  #pragma unroll
  for (int f = 0; f < 4; ++f) oacc[f] = (v8f){0,0,0,0,0,0,0,0};
  float mrow[8], lrow[8];
  #pragma unroll
  for (int r = 0; r < 8; ++r) { mrow[r] = -3.0e38f; lrow[r] = 0.f; }

  for (int kt = 0; kt < SEQ; kt += 64) {
    #pragma unroll
    for (int c = 0; c < 4; ++c) {                    // stage K (async copy) + V (transposed)
      int linear = tid + 128*c;
      int key = linear >> 3, hd8 = (linear & 7) << 3;
#if HAVE_ASYNC
      ASYNC_CP_B128(Ks + key*LDK + hd8,
                    Kh + headoff + (size_t)(kt+key)*HD + hd8);
#else
      *(v8bf*)(Ks + key*LDK + hd8) =
          *(const v8bf*)(Kh + headoff + (size_t)(kt+key)*HD + hd8);
#endif
      v8bf vv = *(const v8bf*)(Vh + headoff + (size_t)(kt+key)*HD + hd8);
      #pragma unroll
      for (int j = 0; j < 8; ++j) Vst[(hd8+j)*LDK + key] = vv[j];
    }
#if HAVE_ASYNC
    ASYNC_WAIT();
#endif
    __syncthreads();

    v8f s[4];
    #pragma unroll
    for (int f = 0; f < 4; ++f) {                    // S = Q K^T  (8 WMMAs)
      const bf16_t* krow = Ks + (f*16 + l15)*LDK;
      s[f] = (v8f){0,0,0,0,0,0,0,0};
      s[f] = wmma_bf16(aq0, frag_from(krow,      g*8, 16 + g*8), s[f]);
      s[f] = wmma_bf16(aq1, frag_from(krow, 32 + g*8, 48 + g*8), s[f]);
    }

    float p[4][8];                                   // online softmax
    #pragma unroll
    for (int r = 0; r < 8; ++r) {
      float tmax = s[0][r];
      #pragma unroll
      for (int f = 1; f < 4; ++f) tmax = fmaxf(tmax, s[f][r]);
      #pragma unroll
      for (int m = 1; m < 16; m <<= 1) tmax = fmaxf(tmax, __shfl_xor(tmax, m, 32));
      float mnew = fmaxf(mrow[r], tmax);
      float cf = __expf(mrow[r] - mnew);
      mrow[r] = mnew; lrow[r] *= cf;
      #pragma unroll
      for (int f = 0; f < 4; ++f) oacc[f][r] *= cf;
      float tsum = 0.f;
      #pragma unroll
      for (int f = 0; f < 4; ++f) { float e = __expf(s[f][r]-mnew); p[f][r]=e; tsum+=e; }
      #pragma unroll
      for (int m = 1; m < 16; m <<= 1) tsum += __shfl_xor(tsum, m, 32);
      lrow[r] += tsum;
    }

    bf16_t* pw = Ps + wave*16*LDK;                   // C-layout -> A-layout via LDS
    #pragma unroll
    for (int f = 0; f < 4; ++f)
      #pragma unroll
      for (int r = 0; r < 8; ++r)
        pw[(r + 8*g)*LDK + f*16 + l15] = (bf16_t)p[f][r];
    asm volatile("s_wait_dscnt 0" ::: "memory");     // wave-local LDS RAW

    const bf16_t* prow = pw + l15*LDK;
    v16bf ap0 = frag_from(prow,      g*8, 16 + g*8);
    v16bf ap1 = frag_from(prow, 32 + g*8, 48 + g*8);
    #pragma unroll
    for (int f = 0; f < 4; ++f) {                    // O += P V  (8 WMMAs)
      const bf16_t* vrow = Vst + (f*16 + l15)*LDK;
      oacc[f] = wmma_bf16(ap0, frag_from(vrow,      g*8, 16 + g*8), oacc[f]);
      oacc[f] = wmma_bf16(ap1, frag_from(vrow, 32 + g*8, 48 + g*8), oacc[f]);
    }
    __syncthreads();
  }

  #pragma unroll
  for (int r = 0; r < 8; ++r) {                      // normalize, store [B,N,D] bf16
    float inv = 1.0f / lrow[r];
    int n = q0 + wave*16 + r + 8*g;
    size_t rowbase = ((size_t)b*SEQ + n)*D_MODEL + h*HD;
    #pragma unroll
    for (int f = 0; f < 4; ++f)
      O[rowbase + f*16 + l15] = (bf16_t)(oacc[f][r] * inv);
  }
}

// ---------------------------------------------------------------------------
// Kernel 3: out = O @ w_proj + b_proj (fp32 out); 256x128 tile, async A staging
// ---------------------------------------------------------------------------
__global__ __launch_bounds__(256)
void proj_kernel(const bf16_t* __restrict__ A, const float* __restrict__ w,
                 const float* __restrict__ bias, float* __restrict__ out)
{
  __shared__ __align__(16) bf16_t As[256*LDA];
  __shared__ __align__(16) bf16_t Bs[128*LDA];
  const int tid = threadIdx.x, wave = tid >> 5, lane = tid & 31;
  const int l15 = lane & 15, g = lane >> 4;
  const int m0 = blockIdx.x * 256, n0 = blockIdx.y * 128;

  v8f acc[2][8];
  #pragma unroll
  for (int i = 0; i < 2; ++i)
    #pragma unroll
    for (int j = 0; j < 8; ++j) acc[i][j] = (v8f){0,0,0,0,0,0,0,0};

  for (int k0 = 0; k0 < D_MODEL; k0 += 32) {
    #pragma unroll
    for (int c = 0; c < 4; ++c) {                    // A (bf16): async straight copy
      int linear = tid + 256*c;                      // 1024 x b128 chunks
      int row = linear >> 2, kk = (linear & 3) << 3;
#if HAVE_ASYNC
      ASYNC_CP_B128(As + row*LDA + kk,
                    A + (size_t)(m0+row)*D_MODEL + k0 + kk);
#else
      *(v8bf*)(As + row*LDA + kk) =
          *(const v8bf*)(A + (size_t)(m0+row)*D_MODEL + k0 + kk);
#endif
    }
    #pragma unroll
    for (int c = 0; c < 4; ++c) {                    // B fp32 -> col-major bf16
      int linear = tid + 256*c;
      int kk = linear >> 5, col = (linear & 31) << 2;
      const float4 f = *(const float4*)(w + (size_t)(k0+kk)*D_MODEL + n0 + col);
      Bs[(col+0)*LDA + kk] = (bf16_t)f.x;
      Bs[(col+1)*LDA + kk] = (bf16_t)f.y;
      Bs[(col+2)*LDA + kk] = (bf16_t)f.z;
      Bs[(col+3)*LDA + kk] = (bf16_t)f.w;
    }
#if HAVE_ASYNC
    ASYNC_WAIT();
#endif
    __syncthreads();
    const bf16_t* arow0 = As + (wave*32 + l15)*LDA;
    v16bf a0 = frag_from(arow0,          g*8, 16 + g*8);
    v16bf a1 = frag_from(arow0 + 16*LDA, g*8, 16 + g*8);
    #pragma unroll
    for (int j = 0; j < 8; ++j) {
      const bf16_t* brow = Bs + (j*16 + l15)*LDA;
      v16bf bfrag = frag_from(brow, g*8, 16 + g*8);
      acc[0][j] = wmma_bf16(a0, bfrag, acc[0][j]);
      acc[1][j] = wmma_bf16(a1, bfrag, acc[1][j]);
    }
    __syncthreads();
  }

  #pragma unroll
  for (int i = 0; i < 2; ++i)
    #pragma unroll
    for (int j = 0; j < 8; ++j) {
      int col = n0 + j*16 + l15;
      float bv = bias[col];
      #pragma unroll
      for (int r = 0; r < 8; ++r) {
        int row = m0 + wave*32 + i*16 + r + 8*g;
        out[(size_t)row*D_MODEL + col] = acc[i][j][r] + bv;
      }
    }
}

// ---------------------------------------------------------------------------
extern "C" void kernel_launch(void* const* d_in, const int* in_sizes, int n_in,
                              void* d_out, int out_size, void* d_ws, size_t ws_size,
                              hipStream_t stream)
{
  const float* x      = (const float*)d_in[0];
  const float* w_qkv  = (const float*)d_in[1];
  const float* w_proj = (const float*)d_in[2];
  const float* b_proj = (const float*)d_in[3];
  float* out = (float*)d_out;

  const size_t elems = (size_t)BATCH*NH*SEQ*HD;      // 8M per tensor
  bf16_t* Qh = (bf16_t*)d_ws;
  bf16_t* Kh = Qh + elems;
  bf16_t* Vh = Kh + elems;
  bf16_t* O  = Vh + elems;                           // total 64 MB bf16

  qkv_kernel <<<dim3(ROWS/256, 3072/128),   256, 0, stream>>>(x, w_qkv, Qh, Kh, Vh);
  attn_kernel<<<dim3(BATCH*NH, SEQ/64),     128, 0, stream>>>(Qh, Kh, Vh, O);
  proj_kernel<<<dim3(ROWS/256, D_MODEL/128),256, 0, stream>>>(O, w_proj, b_proj, out);
}
